// ConditionalSplineFlow_46179488367096
// MI455X (gfx1250) — matrix operand order repaired and checked by reading, not compiled, not run on hardware
//
#include <hip/hip_runtime.h>
#include <hip/hip_bf16.h>

// ---------------------------------------------------------------------------
// Fused conditional spline flow for gfx1250 (MI455X).
// One 256-thread (8 x wave32) workgroup processes 32 batch rows end-to-end:
// all 3 GEMMs per layer run on v_wmma_f32_16x16x32_bf16; the 193MB/layer
// params intermediate lives entirely in the 320KB WGP LDS (never hits HBM).
// ---------------------------------------------------------------------------

namespace {

constexpr int kD       = 64;                 // latent dim
constexpr int kC       = 128;                // context dim
constexpr int kL       = 5;                  // layers
constexpr int kBins    = 8;
constexpr int kH       = 256;                // hidden
constexpr int kP       = kD * (3 * kBins - 1);   // 1472 params per row
constexpr int kTri     = kD * (kD - 1) / 2;      // 2016
constexpr int kRows    = 32;                 // batch rows per workgroup
constexpr int kThreads = 256;

constexpr float kTB   = 5.0f;
constexpr float kMinW = 0.001f;
constexpr float kMinH = 0.001f;
constexpr float kMinD = 0.001f;

// ---- dynamic LDS layout (bytes); total 278784 < 320KB per WGP ----
constexpr int OFF_X   = 0;                        // 32x64  f32   8192
constexpr int OFF_Y   = OFF_X   + kRows * kD * 4; // 32x64  f32   8192
constexpr int OFF_LD  = OFF_Y   + kRows * kD * 4; // logdet + scalar (256B)
constexpr int OFF_CTX = OFF_LD  + 256;            // 32x128 bf16  8192
constexpr int OFF_H1  = OFF_CTX + kRows * kC * 2; // 32x256 bf16 16384
constexpr int OFF_H2  = OFF_H1  + kRows * kH * 2; // 32x256 bf16 16384
constexpr int OFF_WS  = OFF_H2  + kRows * kH * 2; // ---- overlapped region
constexpr int OFF_PAR = OFF_WS;                   // 32x1472 f32 188416 (ph3+)
constexpr int OFF_W3  = OFF_WS  + kRows * kP * 4; // 64x256 bf16  32768 (ph3)
// W1t/W2t staging (<=128KB, phases 1-2) aliases OFF_WS as well.
constexpr int LDS_BYTES = OFF_W3 + 64 * kH * 2;   // 278784

typedef __attribute__((ext_vector_type(16))) __bf16 v16bf;
typedef __attribute__((ext_vector_type(8)))  float  v8f;

union BFrag { v16bf v; uint4 q[2]; };
union Acc   { v8f   v; float f[8]; };

__device__ __forceinline__ unsigned short f2bf(float x) {
  unsigned int u = __float_as_uint(x);                 // round-to-nearest-even
  return (unsigned short)((u + 0x7fffu + ((u >> 16) & 1u)) >> 16);
}
__device__ __forceinline__ float softplusf(float v) {
  return v > 20.f ? v : log1pf(expf(v));
}

// K-loop over one 16x16 output tile. sA is bf16 [M][K] row-major (A layout:
// lane 0-15 row, half-wave selects K base 0/8, two 16B chunks per kstep).
// sB is bf16 [N][K] (B layout: lane 0-15 col, half-wave selects K base 0/8,
// two 16B chunks per kstep, same K order as A).
__device__ __forceinline__ void wmma_kloop(const unsigned short* __restrict__ aRow,
                                           const unsigned short* __restrict__ bCol,
                                           int ksteps, int lhalf, Acc& acc) {
  for (int ks = 0; ks < ksteps; ++ks) {
    BFrag a, b;
    const int ka = ks * 32 + lhalf * 8;
    a.q[0] = *(const uint4*)(aRow + ka);
    a.q[1] = *(const uint4*)(aRow + ka + 16);
    const int kb = ks * 32 + lhalf * 8;
    b.q[0] = *(const uint4*)(bCol + kb);
    b.q[1] = *(const uint4*)(bCol + kb + 16);
    acc.v = __builtin_amdgcn_wmma_f32_16x16x32_bf16(
        false, a.v, false, b.v, (short)0, acc.v, false, false);
  }
}

__global__ __launch_bounds__(kThreads)
void flow_fused_kernel(const float* __restrict__ xin, const float* __restrict__ ctx,
                       const float* __restrict__ W1,  const float* __restrict__ b1,
                       const float* __restrict__ W2,  const float* __restrict__ b2,
                       const float* __restrict__ W3,  const float* __restrict__ b3,
                       const float* __restrict__ lub, const float* __restrict__ lowe,
                       const float* __restrict__ uppe,const float* __restrict__ udia,
                       float* __restrict__ out) {
  extern __shared__ char smem[];
  float*          sX   = (float*)(smem + OFF_X);
  float*          sY   = (float*)(smem + OFF_Y);
  float*          sLd  = (float*)(smem + OFF_LD);
  unsigned short* sCtx = (unsigned short*)(smem + OFF_CTX);
  unsigned short* sH1  = (unsigned short*)(smem + OFF_H1);
  unsigned short* sH2  = (unsigned short*)(smem + OFF_H2);
  unsigned short* sW12 = (unsigned short*)(smem + OFF_WS);   // phases 1-2
  float*          sPar = (float*)(smem + OFF_PAR);           // phase 3+
  unsigned short* sW3  = (unsigned short*)(smem + OFF_W3);   // phase 3

  const int tid   = threadIdx.x;
  const int lane  = tid & 31;
  const int wave  = tid >> 5;
  const int l15   = lane & 15;
  const int lhalf = lane >> 4;
  const int r0    = blockIdx.x * kRows;

  // stage x (f32) and context (bf16, [row][K] layout for WMMA A fragments)
  for (int e = tid; e < kRows * kD; e += kThreads) sX[e] = xin[r0 * kD + e];
  for (int e = tid; e < kRows * kC; e += kThreads) sCtx[e] = f2bf(ctx[r0 * kC + e]);
  if (tid < kRows) sLd[tid] = 0.f;
  __syncthreads();

  for (int l = 0; l < kL; ++l) {
    // ---------------- stage W1^T as bf16 [N=256][K=128] -------------------
    const float* w1 = W1 + l * kC * kH;
    for (int e = tid; e < kC * kH; e += kThreads) {
      int k = e >> 8, n = e & 255;
      sW12[n * kC + k] = f2bf(w1[k * kH + n]);
    }
    __syncthreads();
    // ---------------- GEMM1: hh1 = relu(ctx @ W1 + b1) --------------------
    for (int i = 0; i < 4; ++i) {
      int t = wave * 4 + i, mt = t >> 4, nt = t & 15;
      Acc acc;
      #pragma unroll
      for (int r = 0; r < 8; ++r) acc.f[r] = 0.f;
      wmma_kloop(sCtx + (mt * 16 + l15) * kC, sW12 + (nt * 16 + l15) * kC,
                 kC / 32, lhalf, acc);
      int col = nt * 16 + l15;
      float bv = b1[l * kH + col];
      #pragma unroll
      for (int r = 0; r < 8; ++r) {                 // D layout: VGPR r -> row
        int row = mt * 16 + lhalf * 8 + r;
        sH1[row * kH + col] = f2bf(fmaxf(acc.f[r] + bv, 0.f));
      }
    }
    __syncthreads();
    // ---------------- stage W2^T as bf16 [256][256] -----------------------
    const float* w2 = W2 + l * kH * kH;
    for (int e = tid; e < kH * kH; e += kThreads) {
      int k = e >> 8, n = e & 255;
      sW12[n * kH + k] = f2bf(w2[k * kH + n]);
    }
    __syncthreads();
    // ---------------- GEMM2: hh2 = relu(hh1 @ W2 + b2) --------------------
    for (int i = 0; i < 4; ++i) {
      int t = wave * 4 + i, mt = t >> 4, nt = t & 15;
      Acc acc;
      #pragma unroll
      for (int r = 0; r < 8; ++r) acc.f[r] = 0.f;
      wmma_kloop(sH1 + (mt * 16 + l15) * kH, sW12 + (nt * 16 + l15) * kH,
                 kH / 32, lhalf, acc);
      int col = nt * 16 + l15;
      float bv = b2[l * kH + col];
      #pragma unroll
      for (int r = 0; r < 8; ++r) {
        int row = mt * 16 + lhalf * 8 + r;
        sH2[row * kH + col] = f2bf(fmaxf(acc.f[r] + bv, 0.f));
      }
    }
    __syncthreads();
    // ------- GEMM3: params = hh2 @ W3 + b3, streamed in 64-col chunks -----
    const float* w3 = W3 + l * kH * kP;
    for (int c = 0; c < kP / 64; ++c) {             // 23 chunks
      for (int e = tid; e < 64 * kH; e += kThreads) {
        int k = e >> 6, nl = e & 63;
        sW3[nl * kH + k] = f2bf(w3[k * kP + c * 64 + nl]);
      }
      __syncthreads();
      int mt = wave >> 2, ntl = wave & 3;           // 8 tiles / 8 waves
      Acc acc;
      #pragma unroll
      for (int r = 0; r < 8; ++r) acc.f[r] = 0.f;
      wmma_kloop(sH2 + (mt * 16 + l15) * kH, sW3 + (ntl * 16 + l15) * kH,
                 kH / 32, lhalf, acc);
      int colg = c * 64 + ntl * 16 + l15;
      float bv = b3[l * kP + colg];
      #pragma unroll
      for (int r = 0; r < 8; ++r) {
        int row = mt * 16 + lhalf * 8 + r;
        sPar[row * kP + colg] = acc.f[r] + bv;
      }
      __syncthreads();
    }
    // prefetch next layer's first weight matrix into GL2 while we do VALU work
    if (l + 1 < kL) {
      const char* nw = (const char*)(W1 + (l + 1) * kC * kH);
      __builtin_prefetch(nw + tid * 64, 0, 1);
    }
    // ---------------- rational-quadratic spline (elementwise) -------------
    for (int it = 0; it < (kRows * kD) / kThreads; ++it) {
      int item = tid + it * kThreads;
      int row = item >> 6, dim = item & 63;
      const float* pr = sPar + row * kP + dim * (3 * kBins - 1);
      float uw[kBins], uh[kBins], ud[kBins - 1];
      #pragma unroll
      for (int i = 0; i < kBins; ++i) { uw[i] = pr[i]; uh[i] = pr[kBins + i]; }
      #pragma unroll
      for (int i = 0; i < kBins - 1; ++i) ud[i] = pr[2 * kBins + i];
      float x0 = sX[item];
      bool inside = (x0 >= -kTB) && (x0 <= kTB);
      float xc = fminf(fmaxf(x0, -kTB), kTB);
      float mw = uw[0], mh = uh[0];
      #pragma unroll
      for (int i = 1; i < kBins; ++i) { mw = fmaxf(mw, uw[i]); mh = fmaxf(mh, uh[i]); }
      float ew[kBins], eh[kBins], sw = 0.f, sh = 0.f;
      #pragma unroll
      for (int i = 0; i < kBins; ++i) {
        ew[i] = expf(uw[i] - mw); sw += ew[i];
        eh[i] = expf(uh[i] - mh); sh += eh[i];
      }
      const float wsc = (1.f - kMinW * kBins) / sw;
      const float hsc = (1.f - kMinH * kBins) / sh;
      float cw = -kTB, ch = -kTB;
      float in_cw = -kTB, in_w = 1.f, in_ch = -kTB, in_h = 1.f;
      float in_d0 = 1.f, in_d1 = 1.f;
      #pragma unroll
      for (int i = 0; i < kBins; ++i) {             // running-cumsum bin select
        float wi = (kMinW + wsc * ew[i]) * (2.f * kTB);
        float hi = (kMinH + hsc * eh[i]) * (2.f * kTB);
        float d0 = (i == 0)         ? 1.f : (kMinD + softplusf(ud[i - 1]));
        float d1 = (i == kBins - 1) ? 1.f : (kMinD + softplusf(ud[i]));
        if (xc >= cw) { in_cw = cw; in_w = wi; in_ch = ch; in_h = hi; in_d0 = d0; in_d1 = d1; }
        cw += wi; ch += hi;
      }
      float th    = (xc - in_cw) / in_w;
      float tomt  = th * (1.f - th);
      float delta = in_h / in_w;
      float numer = in_h * (delta * th * th + in_d0 * tomt);
      float denom = delta + (in_d0 + in_d1 - 2.f * delta) * tomt;
      float yv    = in_ch + numer / denom;
      float dnum  = delta * delta *
                    (in_d1 * th * th + 2.f * delta * tomt + in_d0 * (1.f - th) * (1.f - th));
      float ld    = logf(dnum) - 2.f * logf(denom);
      sX[item] = inside ? yv : x0;
      atomicAdd(&sLd[row], inside ? ld : 0.f);      // ds_add_f32
    }
    // ---------------- LU linear: y = L (U x) + b --------------------------
    if (tid == 0) {                                  // scalar log|det U|
      float s = 0.f;
      for (int d2 = 0; d2 < kD; ++d2) s += logf(softplusf(udia[l * kD + d2]) + 0.001f);
      sLd[kRows] = s;
    }
    __syncthreads();                                 // spline done, sLd scalar ready
    const float* up = uppe + l * kTri;
    for (int it = 0; it < (kRows * kD) / kThreads; ++it) {
      int item = tid + it * kThreads;
      int row = item >> 6, j = item & 63;
      float dg = softplusf(udia[l * kD + j]) + 0.001f;
      float a = dg * sX[row * kD + j];
      int base = 63 * j - (j * (j - 1)) / 2;         // triu_indices(D,1) row offset
      for (int k = j + 1; k < kD; ++k) a += up[base + (k - j - 1)] * sX[row * kD + k];
      sY[item] = a;
    }
    __syncthreads();
    const float* lo = lowe + l * kTri;
    for (int it = 0; it < (kRows * kD) / kThreads; ++it) {
      int item = tid + it * kThreads;
      int row = item >> 6, i2 = item & 63;
      float a = sY[item] + lub[l * kD + i2];
      int base = (i2 * (i2 - 1)) / 2;                // tril_indices(D,-1) row offset
      for (int j2 = 0; j2 < i2; ++j2) a += lo[base + j2] * sY[row * kD + j2];
      sX[item] = a;
    }
    if (tid < kRows) sLd[tid] += sLd[kRows];
    __syncthreads();
  }

  // ---------------- base log-prob + accumulated logdet --------------------
  if (tid < kRows) {
    float s = 0.f;
    for (int d2 = 0; d2 < kD; ++d2) { float v = sX[tid * kD + d2]; s += v * v; }
    out[r0 + tid] = -0.5f * s - 58.8120661f + sLd[tid];  // 0.5*64*ln(2*pi)
  }
}

} // namespace

extern "C" void kernel_launch(void* const* d_in, const int* in_sizes, int n_in,
                              void* d_out, int out_size, void* d_ws, size_t ws_size,
                              hipStream_t stream) {
  const float* xin  = (const float*)d_in[0];
  const float* ctx  = (const float*)d_in[1];
  const float* W1   = (const float*)d_in[2];
  const float* b1   = (const float*)d_in[3];
  const float* W2   = (const float*)d_in[4];
  const float* b2   = (const float*)d_in[5];
  const float* W3   = (const float*)d_in[6];
  const float* b3   = (const float*)d_in[7];
  const float* lub  = (const float*)d_in[8];
  const float* lowe = (const float*)d_in[9];
  const float* uppe = (const float*)d_in[10];
  const float* udia = (const float*)d_in[11];
  float* out = (float*)d_out;

  const int batch = in_sizes[0] / kD;       // 32768
  dim3 grid(batch / kRows);                 // 1024 workgroups
  flow_fused_kernel<<<grid, kThreads, LDS_BYTES, stream>>>(
      xin, ctx, W1, b1, W2, b2, W3, b3, lub, lowe, uppe, udia, out);
}
